// RecurrentAutoencoder_46943992545832
// MI455X (gfx1250) — compile-verified, matrix operand order checked by reading, not run on hardware
//
#include <hip/hip_runtime.h>
#include <math.h>

// ---------------------------------------------------------------------------
// RecurrentAutoencoder for MI455X (gfx1250, wave32, WMMA f16)
//   B=64, S=256, V=8000, H=256, E=4
// Key wins: layer-1 forward scan is dead code in the reference (skipped);
// decoder per-step logits ARE the output logits (written once to d_out).
// All recurrent GEMMs run on v_wmma_f32_16x16x32_f16 with f16 state/weights.
// ---------------------------------------------------------------------------

typedef __attribute__((ext_vector_type(16))) _Float16 v16h;
typedef __attribute__((ext_vector_type(8)))  _Float16 v8h;
typedef __attribute__((ext_vector_type(8)))  float    v8f;

constexpr int cB = 64;     // batch
constexpr int cS = 256;    // seq
constexpr int cV = 8000;   // vocab
constexpr int cH = 256;    // hidden
constexpr int cE = 4;      // char embed
constexpr int cG = 4 * cH; // 1024 gate width

__device__ __forceinline__ float sigm(float x) { return 1.0f / (1.0f + __expf(-x)); }

// Load a 16x32 f16 fragment (A layout; B uses the symmetric col-major-by-lane
// layout, identical indexing with row0 = n0). lane%16 -> row, lane/16 -> K half.
// VGPR pairs: e=0..7 -> k = klo+e ; e=8..15 -> k = klo+16+(e-8).
__device__ __forceinline__ v16h load_frag(const _Float16* __restrict__ base,
                                          int ld, int row0, int k0) {
  const int lane = threadIdx.x & 31;
  const int r   = row0 + (lane & 15);
  const int klo = k0 + ((lane >> 4) << 3);
  const _Float16* p = base + (size_t)r * ld + klo;
  v8h lo = *(const v8h*)(p);       // 16B aligned: ld multiple of 32, k0 mult 32
  v8h hi = *(const v8h*)(p + 16);
  v16h out;
#pragma unroll
  for (int e = 0; e < 8; ++e) { out[e] = lo[e]; out[e + 8] = hi[e]; }
  return out;
}

// ------------------------------- prep kernels ------------------------------

__global__ void k_cast_f16(_Float16* __restrict__ dst, const float* __restrict__ src, int n) {
  int i = blockIdx.x * blockDim.x + threadIdx.x;
  if (i < n) dst[i] = (_Float16)src[i];
}

__global__ void k_fill_f16(_Float16* __restrict__ p, int n) {
  int i = blockIdx.x * blockDim.x + threadIdx.x;
  if (i < n) p[i] = (_Float16)0.0f;
}

__global__ void k_fill_f32(float* __restrict__ p, int n) {
  int i = blockIdx.x * blockDim.x + threadIdx.x;
  if (i < n) p[i] = 0.0f;
}

__global__ void k_copy_f32(float* __restrict__ dst, const float* __restrict__ src, int n) {
  int i = blockIdx.x * blockDim.x + threadIdx.x;
  if (i < n) dst[i] = src[i];
}

// xe[(s*B+b)*E+e] = embed[x[b*S+s]*E+e]
__global__ void k_embed_gather(float* __restrict__ xe, const int* __restrict__ x,
                               const float* __restrict__ embed) {
  int i = blockIdx.x * blockDim.x + threadIdx.x;
  if (i >= cS * cB * cE) return;
  int e  = i & (cE - 1);
  int sb = i >> 2;
  int b  = sb & (cB - 1);
  int s  = sb >> 6;
  xe[i] = embed[(size_t)x[b * cS + s] * cE + e];
}

// ------------------------------- LSTM step ---------------------------------
// One kernel = one time step.  grid (B/16=4, H/16=16), block 128 (4 waves).
// Wave w computes the 16x16 gate tile for gate w (i,f,g,o) of this
// (batch-tile, hidden-slice): G = h_in @ whh^T [+ xA @ wih^T].  Tiles land in
// LDS; fused epilogue adds bias + tiny-E input projection and runs the cell.
__global__ void k_lstm_step(
    const _Float16* __restrict__ h_in,   // (64,256) f16  (double-buffered)
    _Float16*       __restrict__ h_out,  // (64,256) f16
    float*          __restrict__ c,      // (64,256) f32  in-place
    const _Float16* __restrict__ whh,    // (1024,256) f16 row-major (n,k)
    const _Float16* __restrict__ xA,     // (64,KX) f16 this step's input, or null
    const _Float16* __restrict__ wih16,  // (1024,KX) f16, or null
    int KX,
    const float*    __restrict__ xs,     // (64,E) f32 tiny input, or null
    const float*    __restrict__ wih_s,  // (1024,E) f32
    int Esmall,
    const float*    __restrict__ bias,   // (1024,)
    _Float16*       __restrict__ ybase,  // out0 row base for this step, or null
    int ycol0,
    float*          __restrict__ hf32)   // (64,256) f32 shadow, or null
{
  __shared__ float gt[4][16][16];
  const int tid  = threadIdx.x;
  const int w    = tid >> 5;          // gate index 0..3
  const int lane = tid & 31;
  const int m0   = blockIdx.x << 4;   // batch tile
  const int h0   = blockIdx.y << 4;   // hidden slice
  const int n0   = w * cH + h0;       // gate column tile

  v8f acc = {};
#pragma unroll
  for (int kt = 0; kt < cH; kt += 32) {
    v16h a = load_frag(h_in, cH, m0, kt);
    v16h b = load_frag(whh,  cH, n0, kt);
    acc = __builtin_amdgcn_wmma_f32_16x16x32_f16(false, a, false, b,
                                                 (short)0, acc, false, false);
  }
  for (int kt = 0; kt < KX; kt += 32) {     // layer-1 input projection (K=512)
    v16h a = load_frag(xA,    KX, m0, kt);
    v16h b = load_frag(wih16, KX, n0, kt);
    acc = __builtin_amdgcn_wmma_f32_16x16x32_f16(false, a, false, b,
                                                 (short)0, acc, false, false);
  }

  const int mb = (lane >> 4) << 3;          // D layout: m = r + 8*(lane/16)
  const int nn = lane & 15;
#pragma unroll
  for (int r = 0; r < 8; ++r) gt[w][r + mb][nn] = acc[r];
  __syncthreads();

  for (int e = tid; e < 256; e += 128) {
    const int m = e >> 4, hc = e & 15;
    const int gb = m0 + m, gh = h0 + hc;
    float gi = gt[0][m][hc] + bias[gh];
    float gf = gt[1][m][hc] + bias[cH + gh];
    float gg = gt[2][m][hc] + bias[2 * cH + gh];
    float go = gt[3][m][hc] + bias[3 * cH + gh];
    if (Esmall > 0) {
      for (int q = 0; q < Esmall; ++q) {
        const float xv = xs[gb * Esmall + q];
        gi += xv * wih_s[(size_t)(gh)            * Esmall + q];
        gf += xv * wih_s[(size_t)(cH + gh)       * Esmall + q];
        gg += xv * wih_s[(size_t)(2 * cH + gh)   * Esmall + q];
        go += xv * wih_s[(size_t)(3 * cH + gh)   * Esmall + q];
      }
    }
    const int si = gb * cH + gh;
    const float cn = sigm(gf) * c[si] + sigm(gi) * tanhf(gg);
    const float hn = sigm(go) * tanhf(cn);
    c[si] = cn;
    h_out[si] = (_Float16)hn;
    if (hf32)  hf32[si] = hn;
    if (ybase) ybase[(size_t)gb * (2 * cH) + ycol0 + gh] = (_Float16)hn;
  }
}

// ------------------------------ decoder logits -----------------------------
// logits[b, s, :] = h @ lin_w^T + lin_b.  grid (4, 125), block 128; each wave
// owns one 16-wide vocab tile, K = 256 -> 8 WMMAs.  Writes directly to d_out.
__global__ void k_dec_logits(const _Float16* __restrict__ h,      // (64,256) f16
                             const _Float16* __restrict__ linw16, // (8000,256) f16
                             const float*    __restrict__ linb,   // (8000,)
                             float* __restrict__ out, int s)
{
  const int tid  = threadIdx.x;
  const int lane = tid & 31;
  const int m0   = blockIdx.x << 4;
  const int nt   = (blockIdx.y << 2) + (tid >> 5);  // 0..499
  const int n0   = nt << 4;

  v8f acc = {};
#pragma unroll
  for (int kt = 0; kt < cH; kt += 32) {
    v16h a = load_frag(h,      cH, m0, kt);
    v16h b = load_frag(linw16, cH, n0, kt);
    acc = __builtin_amdgcn_wmma_f32_16x16x32_f16(false, a, false, b,
                                                 (short)0, acc, false, false);
  }
  const int mb = (lane >> 4) << 3;
  const int v  = n0 + (lane & 15);
#pragma unroll
  for (int r = 0; r < 8; ++r) {
    const int row = m0 + r + mb;
    out[(size_t)row * cS * cV + (size_t)s * cV + v] = acc[r] + linb[v];
  }
}

// --------------------------- decoder argmax+embed --------------------------
// First-occurrence argmax (matches jnp.argmax) over this step's logits row,
// then nxt[b] = embed[idx] (detached greedy feedback).
__global__ void k_dec_argmax(const float* __restrict__ out, int s,
                             const float* __restrict__ embed,
                             float* __restrict__ nxt)
{
  __shared__ float sv[256];
  __shared__ int   si[256];
  const int tid = threadIdx.x;
  const int b   = blockIdx.x;
  const size_t base = (size_t)b * cS * cV + (size_t)s * cV;

  float bv = -INFINITY; int bi = 0x7fffffff;
  for (int v = tid; v < cV; v += 256) {
    const float x = out[base + v];
    if (x > bv) { bv = x; bi = v; }        // strided asc -> keeps first in-thread
  }
  sv[tid] = bv; si[tid] = bi;
  __syncthreads();
  for (int off = 128; off > 0; off >>= 1) {
    if (tid < off) {
      const float ov = sv[tid + off]; const int oi = si[tid + off];
      if (ov > sv[tid] || (ov == sv[tid] && oi < si[tid])) { sv[tid] = ov; si[tid] = oi; }
    }
    __syncthreads();
  }
  if (tid < cE) nxt[b * cE + tid] = embed[(size_t)si[0] * cE + tid];
}

// ------------------------------- launcher ----------------------------------

extern "C" void kernel_launch(void* const* d_in, const int* in_sizes, int n_in,
                              void* d_out, int out_size, void* d_ws, size_t ws_size,
                              hipStream_t stream) {
  (void)in_sizes; (void)n_in; (void)out_size; (void)ws_size;
  const int*   x     = (const int*)  d_in[0];
  const float* embed = (const float*)d_in[1];
  const float* wih0f = (const float*)d_in[2];
  const float* whh0f = (const float*)d_in[3];
  const float* b0f   = (const float*)d_in[4];
  const float* wih0b = (const float*)d_in[5];
  const float* whh0b = (const float*)d_in[6];
  const float* b0b   = (const float*)d_in[7];
  // d_in[8..10] = encoder layer-1 FORWARD weights: yf1 is never consumed by
  // the reference (only h_last/c_last of the backward scan reach the decoder),
  // so that entire scan is skipped.
  const float* wih1b = (const float*)d_in[11];
  const float* whh1b = (const float*)d_in[12];
  const float* b1b   = (const float*)d_in[13];
  const float* dwih  = (const float*)d_in[14];
  const float* dwhh  = (const float*)d_in[15];
  const float* db    = (const float*)d_in[16];
  const float* linw  = (const float*)d_in[17];
  const float* linb  = (const float*)d_in[18];
  float* out = (float*)d_out;

  // workspace carve-up (~25 MB)
  char* wsp = (char*)d_ws;
  auto alloc = [&](size_t bytes) -> void* {
    void* p = (void*)wsp; wsp += (bytes + 255) & ~(size_t)255; return p;
  };
  _Float16* whh0f16 = (_Float16*)alloc(sizeof(_Float16) * cG * cH);
  _Float16* whh0b16 = (_Float16*)alloc(sizeof(_Float16) * cG * cH);
  _Float16* wih1b16 = (_Float16*)alloc(sizeof(_Float16) * cG * 2 * cH);
  _Float16* whh1b16 = (_Float16*)alloc(sizeof(_Float16) * cG * cH);
  _Float16* dwhh16  = (_Float16*)alloc(sizeof(_Float16) * cG * cH);
  _Float16* linw16  = (_Float16*)alloc(sizeof(_Float16) * (size_t)cV * cH);
  float*    xe      = (float*)   alloc(sizeof(float) * cS * cB * cE);
  _Float16* out0    = (_Float16*)alloc(sizeof(_Float16) * (size_t)cS * cB * 2 * cH);
  _Float16* hA      = (_Float16*)alloc(sizeof(_Float16) * cB * cH);
  _Float16* hB      = (_Float16*)alloc(sizeof(_Float16) * cB * cH);
  float*    cbuf    = (float*)   alloc(sizeof(float) * cB * cH);
  float*    hf32    = (float*)   alloc(sizeof(float) * cB * cH);
  float*    nxt     = (float*)   alloc(sizeof(float) * cB * cE);

  auto cast = [&](_Float16* d, const float* s, int n) {
    k_cast_f16<<<(n + 255) / 256, 256, 0, stream>>>(d, s, n);
  };
  cast(whh0f16, whh0f, cG * cH);
  cast(whh0b16, whh0b, cG * cH);
  cast(wih1b16, wih1b, cG * 2 * cH);
  cast(whh1b16, whh1b, cG * cH);
  cast(dwhh16,  dwhh,  cG * cH);
  cast(linw16,  linw,  cV * cH);
  k_embed_gather<<<(cS * cB * cE + 255) / 256, 256, 0, stream>>>(xe, x, embed);

  const dim3 lgrid(cB / 16, cH / 16);   // (4,16)
  auto zero_state = [&] {
    k_fill_f16<<<(cB * cH + 255) / 256, 256, 0, stream>>>(hA, cB * cH);
    k_fill_f32<<<(cB * cH + 255) / 256, 256, 0, stream>>>(cbuf, cB * cH);
  };

  // -------- encoder layer 0 forward: y -> out0[:, :, 0:H] --------
  zero_state();
  for (int t = 0; t < cS; ++t) {
    _Float16* hin  = (t & 1) ? hB : hA;
    _Float16* hout = (t & 1) ? hA : hB;
    k_lstm_step<<<lgrid, 128, 0, stream>>>(
        hin, hout, cbuf, whh0f16,
        nullptr, nullptr, 0,
        xe + (size_t)t * cB * cE, wih0f, cE, b0f,
        out0 + (size_t)t * cB * (2 * cH), 0, nullptr);
  }

  // -------- encoder layer 0 backward: y -> out0[:, :, H:2H] --------
  zero_state();
  for (int t = 0; t < cS; ++t) {
    const int s = cS - 1 - t;
    _Float16* hin  = (t & 1) ? hB : hA;
    _Float16* hout = (t & 1) ? hA : hB;
    k_lstm_step<<<lgrid, 128, 0, stream>>>(
        hin, hout, cbuf, whh0b16,
        nullptr, nullptr, 0,
        xe + (size_t)s * cB * cE, wih0b, cE, b0b,
        out0 + (size_t)s * cB * (2 * cH), cH, nullptr);
  }

  // -------- encoder layer 1 backward: only h_last/c_last survive --------
  zero_state();
  for (int t = 0; t < cS; ++t) {
    const int s = cS - 1 - t;
    _Float16* hin  = (t & 1) ? hB : hA;
    _Float16* hout = (t & 1) ? hA : hB;
    k_lstm_step<<<lgrid, 128, 0, stream>>>(
        hin, hout, cbuf, whh1b16,
        out0 + (size_t)s * cB * (2 * cH), wih1b16, 2 * cH,
        nullptr, nullptr, 0, b1b,
        nullptr, 0, hf32);
  }
  // second reference output: h_last (f32) appended after logits
  k_copy_f32<<<(cB * cH + 255) / 256, 256, 0, stream>>>(
      out + (size_t)cB * cS * cV, hf32, cB * cH);

  // -------- decoder: h/c continue from layer-1-bwd final state --------
  // after 256 steps final h sits in hA (t=255 wrote hA), matching t=0 input.
  k_fill_f32<<<1, 256, 0, stream>>>(nxt, cB * cE);
  for (int t = 0; t < cS; ++t) {
    _Float16* hin  = (t & 1) ? hB : hA;
    _Float16* hout = (t & 1) ? hA : hB;
    k_lstm_step<<<lgrid, 128, 0, stream>>>(
        hin, hout, cbuf, dwhh16,
        nullptr, nullptr, 0,
        nxt, dwih, cE, db,
        nullptr, 0, nullptr);
    k_dec_logits<<<dim3(cB / 16, cV / 64), 128, 0, stream>>>(hout, linw16, linb, out, t);
    k_dec_argmax<<<cB, 256, 0, stream>>>(out, t, embed, nxt);
  }
}
